// Model_17051020165464
// MI455X (gfx1250) — compile-verified, matrix operand order checked by reading, not run on hardware
//
#include <hip/hip_runtime.h>
#include <hip/hip_bf16.h>

typedef __attribute__((ext_vector_type(16))) __bf16 v16bf;
typedef __attribute__((ext_vector_type(8)))  float  v8f;

#define NEG_INF (-3.402823466e+38f)
static constexpr float BN_INV = 0.9999950000374997f; // 1/sqrt(1+1e-5)

// ---------------- conversion / aggregation kernels ----------------
__global__ void k_f32_to_bf16(const float* __restrict__ in, __bf16* __restrict__ out, long n) {
  long i = (long)blockIdx.x * blockDim.x + threadIdx.x;
  if (i < n) out[i] = (__bf16)in[i];
}

__global__ void k_count(const int* __restrict__ dst, float* __restrict__ cnt, int E) {
  int e = blockIdx.x * blockDim.x + threadIdx.x;
  if (e < E) atomicAdd(&cnt[dst[e]], 1.0f);
}

__global__ void k_scatter(const float* __restrict__ x, const int* __restrict__ src,
                          const int* __restrict__ dst, float* __restrict__ agg,
                          long total, int C) {
  long t = (long)blockIdx.x * blockDim.x + threadIdx.x;
  if (t >= total) return;
  int c4 = C >> 2;
  int e  = (int)(t / c4);
  int c  = (int)(t % c4) << 2;
  const float4 v = *(const float4*)(x + (long)src[e] * C + c);
  float* o = agg + (long)dst[e] * C + c;
  atomicAdd(o + 0, v.x);
  atomicAdd(o + 1, v.y);
  atomicAdd(o + 2, v.z);
  atomicAdd(o + 3, v.w);
}

__global__ void k_div_bf16(const float* __restrict__ agg, const float* __restrict__ cnt,
                           __bf16* __restrict__ out, int N, int C) {
  long i = (long)blockIdx.x * blockDim.x + threadIdx.x;
  if (i >= (long)N * C) return;
  int n = (int)(i / C);
  out[i] = (__bf16)(agg[i] / fmaxf(cnt[n], 1.0f));
}

// ---------------- fused dual-WMMA GEMM ----------------
// C[M,N] = A1@W1^T (+ A2@W2^T if HAS2) (+bias), epilogue: 0=none, 1=bn(relu(x)), 2=tanh
// B_KN==0: W is [N,K] row-major (weights); B_KN==1: W is [K,N] row-major (transposed at staging)
// LDS B layout is ALWAYS [n][k] so the fragment-build path is branch-free.
template <int B_KN, int HAS2>
__global__ __launch_bounds__(256)
void k_wmma_gemm(const __bf16* __restrict__ A1, const __bf16* __restrict__ W1,
                 const __bf16* __restrict__ A2, const __bf16* __restrict__ W2,
                 const float* __restrict__ bias,
                 const float* __restrict__ g, const float* __restrict__ bb,
                 float* __restrict__ C, int M, int N, int K, int mode) {
  __shared__ __bf16 lA [64 * 32];                 // [m][k]
  __shared__ __bf16 lB [32 * 32];                 // [n][k]
  __shared__ __bf16 lA2[HAS2 ? 64 * 32 : 1];
  __shared__ __bf16 lB2[HAS2 ? 32 * 32 : 1];

  const int t    = threadIdx.x;
  const int lane = t & 31;
  const int w    = t >> 5;     // 8 waves
  const int wm   = w & 3;      // 4 wave rows (16 each)
  const int wn   = w >> 2;     // 2 wave cols (16 each)
  const int half = lane >> 4;
  const int l15  = lane & 15;
  const int mBase = blockIdx.y * 64;
  const int nBase = blockIdx.x * 32;

  // staging coords: each thread moves 8 contiguous bf16 (16 bytes)
  const int ar_ = t >> 2;          // A row 0..63
  const int ak_ = (t & 3) << 3;    // A k-offset {0,8,16,24}
  const int br_ = (t & 127) >> 2;  // B outer 0..31 (threads 0..127)
  const int bk_ = (t & 3) << 3;    // B inner offset {0,8,16,24}

  v8f acc = {};

  for (int k0 = 0; k0 < K; k0 += 32) {
    const bool fullK = (k0 + 32 <= K);

    // ---- stage A (64x32 [m][k]) ----
    {
      const int  gm   = mBase + ar_;
      const long base = (long)gm * K + k0 + ak_;
      if (fullK && gm < M) {
        *(uint4*)&lA[t * 8] = *(const uint4*)(A1 + base);
        if (HAS2) *(uint4*)&lA2[t * 8] = *(const uint4*)(A2 + base);
        if (k0 + 32 < K) {  // prefetch next K-tile -> global_prefetch
          __builtin_prefetch(A1 + base + 32, 0, 1);
          if (HAS2) __builtin_prefetch(A2 + base + 32, 0, 1);
        }
      } else {
        #pragma unroll
        for (int i = 0; i < 8; ++i) {
          int gk = k0 + ak_ + i;
          bool ok = (gm < M) && (gk < K);
          lA[t * 8 + i] = ok ? A1[(long)gm * K + gk] : (__bf16)0.0f;
          if (HAS2) lA2[t * 8 + i] = ok ? A2[(long)gm * K + gk] : (__bf16)0.0f;
        }
      }
    }

    // ---- stage B (32x32 -> LDS [n][k]) : threads 0..127 ----
    if (t < 128) {
      if (!B_KN) {                              // W[N,K]: contiguous along K, direct copy
        const int  gn   = nBase + br_;          // br_ = n row, bk_ = k offset
        const long base = (long)gn * K + k0 + bk_;
        if (fullK && gn < N) {
          *(uint4*)&lB[t * 8] = *(const uint4*)(W1 + base);
          if (HAS2) *(uint4*)&lB2[t * 8] = *(const uint4*)(W2 + base);
          if (k0 + 32 < K) {
            __builtin_prefetch(W1 + base + 32, 0, 1);
            if (HAS2) __builtin_prefetch(W2 + base + 32, 0, 1);
          }
        } else {
          #pragma unroll
          for (int i = 0; i < 8; ++i) {
            int gk = k0 + bk_ + i;
            bool ok = (gn < N) && (gk < K);
            lB[t * 8 + i] = ok ? W1[(long)gn * K + gk] : (__bf16)0.0f;
            if (HAS2) lB2[t * 8 + i] = ok ? W2[(long)gn * K + gk] : (__bf16)0.0f;
          }
        }
      } else {                                   // W[K,N]: load N-contiguous, transpose into [n][k]
        const int gk  = k0 + br_;                // br_ = k row, bk_ = n offset
        const int gn0 = nBase + bk_;
        __bf16 tmp[8];
        if (gk < K && gn0 + 8 <= N) {
          *(uint4*)tmp = *(const uint4*)(W1 + (long)gk * N + gn0);
        } else {
          #pragma unroll
          for (int i = 0; i < 8; ++i) {
            bool ok = (gk < K) && (gn0 + i < N);
            tmp[i] = ok ? W1[(long)gk * N + gn0 + i] : (__bf16)0.0f;
          }
        }
        #pragma unroll
        for (int i = 0; i < 8; ++i) lB[(bk_ + i) * 32 + br_] = tmp[i];
      }
    }
    __syncthreads();

    // ---- fragments per CDNA5 16-bit A/B layout (branch-free) ----
    // lanes 0-15: VGPR0-3 K=0..7, VGPR4-7 K=16..23; lanes 16-31: +8
    const __bf16* arp = &lA[(wm * 16 + l15) * 32];
    const __bf16* brp = &lB[(wn * 16 + l15) * 32];
    v16bf af, bfv;
    #pragma unroll
    for (int i = 0; i < 8; ++i) {
      int kk = ((i < 4) ? 0 : 16) + half * 8 + ((i & 3) << 1);
      af [2 * i] = arp[kk]; af [2 * i + 1] = arp[kk + 1];
      bfv[2 * i] = brp[kk]; bfv[2 * i + 1] = brp[kk + 1];
    }
    acc = __builtin_amdgcn_wmma_f32_16x16x32_bf16(false, af, false, bfv,
                                                  (short)0, acc, false, false);
    if (HAS2) {
      const __bf16* arp2 = &lA2[(wm * 16 + l15) * 32];
      const __bf16* brp2 = &lB2[(wn * 16 + l15) * 32];
      v16bf af2, bf2;
      #pragma unroll
      for (int i = 0; i < 8; ++i) {
        int kk = ((i < 4) ? 0 : 16) + half * 8 + ((i & 3) << 1);
        af2[2 * i] = arp2[kk]; af2[2 * i + 1] = arp2[kk + 1];
        bf2[2 * i] = brp2[kk]; bf2[2 * i + 1] = brp2[kk + 1];
      }
      acc = __builtin_amdgcn_wmma_f32_16x16x32_bf16(false, af2, false, bf2,
                                                    (short)0, acc, false, false);
    }
    __syncthreads();
  }

  // C/D layout: lanes 0-15 -> M = vgpr, lanes 16-31 -> M = vgpr+8; N = lane&15
  const int row0 = mBase + wm * 16;
  const int col  = nBase + wn * 16 + l15;
  #pragma unroll
  for (int v = 0; v < 8; ++v) {
    int m = row0 + v + half * 8;
    if (m < M && col < N) {
      float val = acc[v];
      if (bias) val += bias[col];
      if (mode == 1) val = g[col] * fmaxf(val, 0.0f) * BN_INV + bb[col];
      else if (mode == 2) val = tanhf(val);
      C[(long)m * N + col] = val;
    }
  }
}

// ---------------- softmax over node axis (per channel) ----------------
__global__ void k_col_max(const float* __restrict__ a, float* __restrict__ mx, int N, int C) {
  __shared__ float s[256];
  int c = blockIdx.x;
  float m = NEG_INF;
  for (int n = threadIdx.x; n < N; n += blockDim.x) m = fmaxf(m, a[(long)n * C + c]);
  s[threadIdx.x] = m;
  __syncthreads();
  for (int st = 128; st > 0; st >>= 1) {
    if (threadIdx.x < st) s[threadIdx.x] = fmaxf(s[threadIdx.x], s[threadIdx.x + st]);
    __syncthreads();
  }
  if (threadIdx.x == 0) mx[c] = s[0];
}

__global__ void k_col_sumexp(const float* __restrict__ a, const float* __restrict__ mx,
                             float* __restrict__ sm, int N, int C) {
  __shared__ float s[256];
  int c = blockIdx.x;
  float m = mx[c], acc = 0.0f;
  for (int n = threadIdx.x; n < N; n += blockDim.x) acc += expf(a[(long)n * C + c] - m);
  s[threadIdx.x] = acc;
  __syncthreads();
  for (int st = 128; st > 0; st >>= 1) {
    if (threadIdx.x < st) s[threadIdx.x] += s[threadIdx.x + st];
    __syncthreads();
  }
  if (threadIdx.x == 0) sm[c] = s[0];
}

__global__ void k_softmax_T(const float* __restrict__ a, const float* __restrict__ mx,
                            const float* __restrict__ sm, __bf16* __restrict__ attT,
                            int N, int C) {
  long i = (long)blockIdx.x * blockDim.x + threadIdx.x;
  if (i >= (long)N * C) return;
  int n = (int)(i / C), c = (int)(i % C);
  attT[(long)c * N + n] = (__bf16)(expf(a[i] - mx[c]) / sm[c]);
}

// ---------------- tiny CNN / FC tail ----------------
__global__ void k_conv3x3(const float* __restrict__ in, const float* __restrict__ wt,
                          const float* __restrict__ bc, const float* __restrict__ g,
                          const float* __restrict__ bb, float* __restrict__ out,
                          int Cin, int Cout, int Hin, int Win, int Hout, int Wout,
                          int stride, int pad, int bnrelu) {
  int idx = blockIdx.x * blockDim.x + threadIdx.x;
  int total = Cout * Hout * Wout;
  if (idx >= total) return;
  int wo = idx % Wout, ho = (idx / Wout) % Hout, co = idx / (Wout * Hout);
  float acc = bc[co];
  for (int ci = 0; ci < Cin; ++ci)
    for (int kh = 0; kh < 3; ++kh) {
      int hi = ho * stride - pad + kh;
      if (hi < 0 || hi >= Hin) continue;
      for (int kw = 0; kw < 3; ++kw) {
        int wi = wo * stride - pad + kw;
        if (wi < 0 || wi >= Win) continue;
        acc += in[((long)ci * Hin + hi) * Win + wi] *
               wt[(((long)co * Cin + ci) * 3 + kh) * 3 + kw];
      }
    }
  if (bnrelu) acc = fmaxf(g[co] * acc * BN_INV + bb[co], 0.0f);  // relu(bn(x))
  out[idx] = acc;
}

__global__ void k_maxpool3(const float* __restrict__ in, float* __restrict__ out,
                           int C, int H, int W) {
  int idx = blockIdx.x * blockDim.x + threadIdx.x;
  if (idx >= C * H * W) return;
  int w = idx % W, h = (idx / W) % H, c = idx / (W * H);
  float m = NEG_INF;
  for (int dh = -1; dh <= 1; ++dh) {
    int hh = h + dh;
    if (hh < 0 || hh >= H) continue;
    for (int dw = -1; dw <= 1; ++dw) {
      int ww = w + dw;
      if (ww < 0 || ww >= W) continue;
      m = fmaxf(m, in[((long)c * H + hh) * W + ww]);
    }
  }
  out[idx] = m;
}

__global__ void k_fc(const float* __restrict__ in, const float* __restrict__ wt,
                     const float* __restrict__ b, float* __restrict__ out,
                     int O, int K, int act) {
  int o = blockIdx.x * blockDim.x + threadIdx.x;
  if (o >= O) return;
  float acc = b[o];
  for (int k = 0; k < K; ++k) acc += in[k] * wt[(long)o * K + k];
  if (act) acc = 1.0f / (1.0f + expf(-acc));
  out[o] = acc;
}

// ---------------- host orchestration ----------------
extern "C" void kernel_launch(void* const* d_in, const int* in_sizes, int n_in,
                              void* d_out, int out_size, void* d_ws, size_t ws_size,
                              hipStream_t stream) {
  const float* x    = (const float*)d_in[0];
  const int*   ei   = (const int*)d_in[1];
  const float* w1l  = (const float*)d_in[2];
  const float* b1   = (const float*)d_in[3];
  const float* w1r  = (const float*)d_in[4];
  const float* bn1g = (const float*)d_in[5];
  const float* bn1b = (const float*)d_in[6];
  const float* w2l  = (const float*)d_in[7];
  const float* b2   = (const float*)d_in[8];
  const float* w2r  = (const float*)d_in[9];
  const float* fa1w = (const float*)d_in[10];
  const float* fa1b = (const float*)d_in[11];
  const float* fa2w = (const float*)d_in[12];
  const float* fa2b = (const float*)d_in[13];
  const float* cw   = (const float*)d_in[14];
  const float* cb   = (const float*)d_in[15];
  const float* ng   = (const float*)d_in[16];
  const float* nb   = (const float*)d_in[17];
  const float* c1w  = (const float*)d_in[18];
  const float* c1b  = (const float*)d_in[19];
  const float* c1g  = (const float*)d_in[20];
  const float* c1bb = (const float*)d_in[21];
  const float* c2w  = (const float*)d_in[22];
  const float* c2b  = (const float*)d_in[23];
  const float* c2g  = (const float*)d_in[24];
  const float* c2bb = (const float*)d_in[25];
  const float* f1w  = (const float*)d_in[26];
  const float* f1b  = (const float*)d_in[27];
  const float* f2w  = (const float*)d_in[28];
  const float* f2b  = (const float*)d_in[29];
  (void)n_in; (void)out_size; (void)ws_size;

  const int Nn = in_sizes[0] / 2560;   // 10000
  const int E  = in_sizes[1] / 2;      // 160000
  const int* src = ei;
  const int* dst = ei + E;

  char* ws = (char*)d_ws;
  size_t off = 0;
  auto alloc = [&](size_t bytes) -> void* {
    void* p = ws + off;
    off = (off + bytes + 255) & ~(size_t)255;
    return p;
  };

  __bf16* xb16  = (__bf16*)alloc((size_t)Nn * 2560 * 2);
  __bf16* w1lb  = (__bf16*)alloc((size_t)1024 * 2560 * 2);
  __bf16* w1rb  = (__bf16*)alloc((size_t)1024 * 2560 * 2);
  float*  agg1f = (float*) alloc((size_t)Nn * 2560 * 4);
  __bf16* agg1b = (__bf16*)alloc((size_t)Nn * 2560 * 2);
  float*  cnt   = (float*) alloc((size_t)Nn * 4);
  __bf16* h1b   = (__bf16*)alloc((size_t)Nn * 1024 * 2);
  __bf16* agg2b = (__bf16*)alloc((size_t)Nn * 1024 * 2);
  __bf16* w2lb  = (__bf16*)alloc((size_t)512 * 1024 * 2);
  __bf16* w2rb  = (__bf16*)alloc((size_t)512 * 1024 * 2);
  float*  h2f   = (float*) alloc((size_t)Nn * 512 * 4);
  __bf16* h2b   = (__bf16*)alloc((size_t)Nn * 512 * 2);
  __bf16* fa1wb = (__bf16*)alloc((size_t)128 * 512 * 2);
  __bf16* fa2wb = (__bf16*)alloc((size_t)128 * 128 * 2);
  float*  a1f   = (float*) alloc((size_t)Nn * 128 * 4);
  __bf16* a1b   = (__bf16*)alloc((size_t)Nn * 128 * 2);
  float*  a2f   = (float*) alloc((size_t)Nn * 128 * 4);
  float*  mx    = (float*) alloc(128 * 4);
  float*  sm    = (float*) alloc(128 * 4);
  __bf16* attTb = (__bf16*)alloc((size_t)128 * Nn * 2);
  float*  embf  = (float*) alloc((size_t)128 * 512 * 4);
  float*  fcmid = (float*) alloc(1024 * 4);
  // aliases into dead regions
  float* h1f   = agg1f;                          // [Nn,1024] (agg1f dead after conversion)
  float* agg2f = agg1f + (size_t)Nn * 1024;      // [Nn,1024]
  float* cnnA  = (float*)xb16;                   // CNN ping (xb16 dead after SAGE1)
  float* cnnB  = cnnA + (size_t)64 * 132 * 516;  // CNN pong

  auto g1 = [](long n) { return dim3((unsigned)((n + 255) / 256)); };

  // ---- bf16 conversion of inputs / weights ----
  long nx = (long)Nn * 2560;
  k_f32_to_bf16<<<g1(nx), 256, 0, stream>>>(x, xb16, nx);
  k_f32_to_bf16<<<g1(1024L * 2560), 256, 0, stream>>>(w1l, w1lb, 1024L * 2560);
  k_f32_to_bf16<<<g1(1024L * 2560), 256, 0, stream>>>(w1r, w1rb, 1024L * 2560);
  k_f32_to_bf16<<<g1(512L * 1024), 256, 0, stream>>>(w2l, w2lb, 512L * 1024);
  k_f32_to_bf16<<<g1(512L * 1024), 256, 0, stream>>>(w2r, w2rb, 512L * 1024);
  k_f32_to_bf16<<<g1(128L * 512), 256, 0, stream>>>(fa1w, fa1wb, 128L * 512);
  k_f32_to_bf16<<<g1(128L * 128), 256, 0, stream>>>(fa2w, fa2wb, 128L * 128);

  // ---- SAGE layer 1: mean aggregation (x fits in 192MB L2 -> gather mostly L2) ----
  hipMemsetAsync(agg1f, 0, (size_t)Nn * 2560 * 4, stream);
  hipMemsetAsync(cnt, 0, (size_t)Nn * 4, stream);
  k_count<<<g1(E), 256, 0, stream>>>(dst, cnt, E);
  long tot1 = (long)E * (2560 / 4);
  k_scatter<<<g1(tot1), 256, 0, stream>>>(x, src, dst, agg1f, tot1, 2560);
  k_div_bf16<<<g1(nx), 256, 0, stream>>>(agg1f, cnt, agg1b, Nn, 2560);
  // h1 = bn(relu(agg@w1l^T + b1 + x@w1r^T))  -- dual-WMMA fused
  k_wmma_gemm<0, 1><<<dim3(1024 / 32, (Nn + 63) / 64), 256, 0, stream>>>(
      agg1b, w1lb, xb16, w1rb, b1, bn1g, bn1b, h1f, Nn, 1024, 2560, 1);
  long nh1 = (long)Nn * 1024;
  k_f32_to_bf16<<<g1(nh1), 256, 0, stream>>>(h1f, h1b, nh1);

  // ---- SAGE layer 2 ----
  hipMemsetAsync(agg2f, 0, (size_t)Nn * 1024 * 4, stream);
  long tot2 = (long)E * (1024 / 4);
  k_scatter<<<g1(tot2), 256, 0, stream>>>(h1f, src, dst, agg2f, tot2, 1024);
  k_div_bf16<<<g1(nh1), 256, 0, stream>>>(agg2f, cnt, agg2b, Nn, 1024);
  k_wmma_gemm<0, 1><<<dim3(512 / 32, (Nn + 63) / 64), 256, 0, stream>>>(
      agg2b, w2lb, h1b, w2rb, b2, nullptr, nullptr, h2f, Nn, 512, 1024, 0);
  long nh2 = (long)Nn * 512;
  k_f32_to_bf16<<<g1(nh2), 256, 0, stream>>>(h2f, h2b, nh2);

  // ---- attention pooling ----
  k_wmma_gemm<0, 0><<<dim3(128 / 32, (Nn + 63) / 64), 256, 0, stream>>>(
      h2b, fa1wb, nullptr, nullptr, fa1b, nullptr, nullptr, a1f, Nn, 128, 512, 2);
  long na = (long)Nn * 128;
  k_f32_to_bf16<<<g1(na), 256, 0, stream>>>(a1f, a1b, na);
  k_wmma_gemm<0, 0><<<dim3(128 / 32, (Nn + 63) / 64), 256, 0, stream>>>(
      a1b, fa2wb, nullptr, nullptr, fa2b, nullptr, nullptr, a2f, Nn, 128, 128, 0);
  k_col_max<<<128, 256, 0, stream>>>(a2f, mx, Nn, 128);
  k_col_sumexp<<<128, 256, 0, stream>>>(a2f, mx, sm, Nn, 128);
  k_softmax_T<<<g1(na), 256, 0, stream>>>(a2f, mx, sm, attTb, Nn, 128);
  // emb[128,512] = attT[128,Nn] @ h2[Nn,512]  (B row-major K x N, transposed at staging)
  k_wmma_gemm<1, 0><<<dim3(512 / 32, (128 + 63) / 64), 256, 0, stream>>>(
      attTb, h2b, nullptr, nullptr, nullptr, nullptr, nullptr, embf, 128, 512, Nn, 0);

  // ---- CNN tail (tiny) ----
  int H = 132, W = 516;
  k_conv3x3<<<g1((long)64 * H * W), 256, 0, stream>>>(
      embf, cw, cb, ng, nb, cnnA, 1, 64, 128, 512, H, W, 1, 3, 1);
  for (int i = 0; i < 5; ++i) {
    int Ho = (H - 1) / 2 + 1, Wo = (W - 1) / 2 + 1;
    k_conv3x3<<<g1((long)64 * Ho * Wo), 256, 0, stream>>>(
        cnnA, c1w, c1b, c1g, c1bb, cnnB, 64, 64, H, W, Ho, Wo, 2, 1, 1);
    k_maxpool3<<<g1((long)64 * Ho * Wo), 256, 0, stream>>>(cnnB, cnnA, 64, Ho, Wo);
    H = Ho; W = Wo;
  }
  int Ho = (H - 1) / 2 + 1, Wo = (W - 1) / 2 + 1;  // 3 x 9
  k_conv3x3<<<g1((long)16 * Ho * Wo), 256, 0, stream>>>(
      cnnA, c2w, c2b, c2g, c2bb, cnnB, 64, 16, H, W, Ho, Wo, 2, 1, 1);
  k_maxpool3<<<g1((long)16 * Ho * Wo), 256, 0, stream>>>(cnnB, cnnA, 16, Ho, Wo);

  // ---- FC head + sigmoid -> d_out [1,14] f32 ----
  k_fc<<<g1(1024), 256, 0, stream>>>(cnnA, f1w, f1b, fcmid, 1024, 16 * Ho * Wo, 0);
  k_fc<<<1, 256, 0, stream>>>(fcmid, f2w, f2b, (float*)d_out, 14, 1024, 1);
}
